// Lstm_81415400063513
// MI455X (gfx1250) — matrix operand order repaired and checked by reading, not compiled, not run on hardware
//
#include <hip/hip_runtime.h>

// ---------------------------------------------------------------------------
// Persistent-LSTM for MI455X (gfx1250, wave32, WMMA).
//   B=2048, T=512, I=5, H=128.  Grid = B/16 workgroups of 256 threads (8 waves).
//   Each wave owns 16 hidden units (all 4 gates); W_hh/W_ih slices live in
//   registers as bf16 WMMA B-fragments.  Bias is folded into the x K-tile
//   (A[k=5]=1.0, B[k=5]=b_ih+b_hh) so each gate's 5-WMMA chain starts from an
//   inline-zero C operand.  h is exchanged through double-buffered LDS (bf16),
//   c stays in registers in WMMA C/D layout.
// ---------------------------------------------------------------------------

typedef __bf16 bf16_t;
typedef __attribute__((ext_vector_type(16))) __bf16 v16bf;
typedef __attribute__((ext_vector_type(8)))  __bf16 v8bf;
typedef __attribute__((ext_vector_type(8)))  float  v8f;

union V16BF { v16bf v; v8bf h[2]; };

#define HID    128
#define INSZ   5
#define TSTEPS 512
#define BTILE  16
#define NWAVES 8
#define LDSROW 136   // 128 + 8 pad: keeps 16B alignment, spreads LDS banks

__device__ __forceinline__ float fast_tanh(float x) {
#if __has_builtin(__builtin_amdgcn_tanhf)
    return __builtin_amdgcn_tanhf(x);
#else
    return 2.0f * __builtin_amdgcn_rcpf(1.0f + __expf(-2.0f * x)) - 1.0f;
#endif
}
__device__ __forceinline__ float fast_sigmoid(float x) {
    // sigmoid(x) = 0.5*tanh(x/2) + 0.5  (1 trans + mul + fma)
    return __builtin_fmaf(0.5f, fast_tanh(0.5f * x), 0.5f);
}

__global__ __launch_bounds__(256, 1)
void lstm_persistent_kernel(const float* __restrict__ x,
                            const float* __restrict__ W_ih,
                            const float* __restrict__ W_hh,
                            const float* __restrict__ b_ih,
                            const float* __restrict__ b_hh,
                            const float* __restrict__ W_lin,
                            const float* __restrict__ b_lin,
                            float* __restrict__ out) {
    __shared__ bf16_t hbuf[2][BTILE * LDSROW];   // double-buffered h tile (bf16)
    __shared__ float  red[NWAVES][BTILE];        // final linear reduction

    const int tid   = threadIdx.x;
    const int wid   = tid >> 5;
    const int lane  = tid & 31;
    const int half  = lane >> 4;      // 0: K-lo / rows 0..7, 1: K-hi / rows 8..15
    const int nl    = lane & 15;
    const int hbase = wid * 16;       // this wave's hidden slice [hbase, hbase+16)
    const int col   = hbase + nl;     // hidden column owned by this lane
    const long b0   = (long)blockIdx.x * BTILE;

    // ---- one-time: W_hh / W_ih slices into register-resident bf16 B-frags.
    // B-frag (32x16 bf16), contiguous k order as in A: lane n<16 -> col n, K rows 0..15 packed in
    // pairs over 8 VGPRs; lanes 16..31 -> K rows 16..31.  B[k][n] = W_hh[n][k],
    // W_hh rows contiguous -> each lane reads 16 contiguous floats (4x float4).
    V16BF Bf[4][5];
    #pragma unroll
    for (int g = 0; g < 4; ++g) {
        const int row = g * HID + col;            // gate-output row
        const float* wr = W_hh + (long)row * HID;
        #pragma unroll
        for (int kt = 0; kt < 4; ++kt) {
            const float4* w4 = (const float4*)(wr + kt * 32 + half * 16);
            #pragma unroll
            for (int q = 0; q < 4; ++q) {
                float4 f = w4[q];
                Bf[g][kt].v[4 * q + 0] = (bf16_t)f.x;
                Bf[g][kt].v[4 * q + 1] = (bf16_t)f.y;
                Bf[g][kt].v[4 * q + 2] = (bf16_t)f.z;
                Bf[g][kt].v[4 * q + 3] = (bf16_t)f.w;
            }
        }
        // kt = 4: x-projection K-tile.  Lane-half 0 holds K=0..15:
        //   K=0..4 -> W_ih row, K=5 -> fused bias (paired with A[k=5]=1.0).
        #pragma unroll
        for (int e = 0; e < 16; ++e) Bf[g][4].v[e] = (bf16_t)0.0f;
        if (half == 0) {
            #pragma unroll
            for (int i = 0; i < INSZ; ++i)
                Bf[g][4].v[i] = (bf16_t)W_ih[row * INSZ + i];
            Bf[g][4].v[INSZ] = (bf16_t)(b_ih[row] + b_hh[row]);
        }
    }

    // x row for this lane's batch row (A-matrix rows live in lanes 0..15)
    const float* xrow = x + (b0 + nl) * (long)(TSTEPS * INSZ);

    // Persistent x A-fragment: only K=0..4 change per step; K=5 = 1.0 (bias).
    V16BF ax;
    #pragma unroll
    for (int e = 0; e < 16; ++e) ax.v[e] = (bf16_t)0.0f;
    if (lane < 16) ax.v[INSZ] = (bf16_t)1.0f;

    // h(-1) = 0
    for (int i = tid; i < BTILE * LDSROW; i += 256) hbuf[0][i] = (bf16_t)0.0f;
    __syncthreads();

    // c state: 16x16 f32 tile in WMMA C/D layout (8 VGPRs/lane)
    v8f cst, hnew;
    #pragma unroll
    for (int e = 0; e < 8; ++e) { cst[e] = 0.0f; hnew[e] = 0.0f; }

    #pragma unroll 2
    for (int t = 0; t < TSTEPS; ++t) {
        const int cur = t & 1;
        const int nxt = cur ^ 1;

        // refresh x elements of the persistent A fragment
        if (lane < 16) {
            const float* xp = xrow + t * INSZ;
            #pragma unroll
            for (int i = 0; i < INSZ; ++i) ax.v[i] = (bf16_t)xp[i];
            if (t + 8 < TSTEPS) __builtin_prefetch(xp + 8 * INSZ, 0, 0);
        }

        const bf16_t* hb = hbuf[cur];
        v8f acc[4];

        // kt = 0: start each gate chain from inline-zero C
        {
            V16BF a;
            const int c0 = half * 16;
            a.h[0] = *(const v8bf*)(hb + nl * LDSROW + c0);
            a.h[1] = *(const v8bf*)(hb + nl * LDSROW + c0 + 8);
            const v8f zacc = {0.0f, 0.0f, 0.0f, 0.0f, 0.0f, 0.0f, 0.0f, 0.0f};
            #pragma unroll
            for (int g = 0; g < 4; ++g)
                acc[g] = __builtin_amdgcn_wmma_f32_16x16x32_bf16(
                    false, a.v, false, Bf[g][0].v, (short)0, zacc, false, false);
        }
        // kt = 1..3: h K-tiles;  kt = 4: x + bias K-tile
        #pragma unroll
        for (int kt = 1; kt < 5; ++kt) {
            V16BF a;
            if (kt < 4) {
                const int c0 = kt * 32 + half * 16;
                a.h[0] = *(const v8bf*)(hb + nl * LDSROW + c0);
                a.h[1] = *(const v8bf*)(hb + nl * LDSROW + c0 + 8);
            } else {
                a = ax;
            }
            #pragma unroll
            for (int g = 0; g < 4; ++g)
                acc[g] = __builtin_amdgcn_wmma_f32_16x16x32_bf16(
                    false, a.v, false, Bf[g][kt].v, (short)0, acc[g],
                    false, false);
        }

        // gate nonlinearities + state update (co-executes with XDL WMMAs)
        #pragma unroll
        for (int e = 0; e < 8; ++e) {
            float ig = fast_sigmoid(acc[0][e]);
            float fg = fast_sigmoid(acc[1][e]);
            float gg = fast_tanh   (acc[2][e]);
            float og = fast_sigmoid(acc[3][e]);
            float cv = fg * cst[e] + ig * gg;
            cst[e]  = cv;
            hnew[e] = og * fast_tanh(cv);
        }

        // publish h(t) (bf16) for all waves; rows e+8*half, fixed column
        bf16_t* ho = hbuf[nxt];
        #pragma unroll
        for (int e = 0; e < 8; ++e)
            ho[(e + half * 8) * LDSROW + col] = (bf16_t)hnew[e];

        __syncthreads();
    }

    // ---- final linear: out[b] = sum_j h[b][j] * W_lin[j] + b_lin
    const float wl = W_lin[col];
    float p[8];
    #pragma unroll
    for (int e = 0; e < 8; ++e) p[e] = hnew[e] * wl;
    // reduce across the 16 lanes of each half (this wave's 16 hidden columns)
    #pragma unroll
    for (int off = 1; off < 16; off <<= 1) {
        #pragma unroll
        for (int e = 0; e < 8; ++e) p[e] += __shfl_xor(p[e], off, 32);
    }
    if (nl == 0) {
        #pragma unroll
        for (int e = 0; e < 8; ++e) red[wid][half * 8 + e] = p[e];
    }
    __syncthreads();
    if (tid < BTILE) {
        float s = b_lin[0];
        #pragma unroll
        for (int w = 0; w < NWAVES; ++w) s += red[w][tid];
        out[b0 + tid] = s;
    }
}

extern "C" void kernel_launch(void* const* d_in, const int* in_sizes, int n_in,
                              void* d_out, int out_size, void* d_ws, size_t ws_size,
                              hipStream_t stream) {
    const float* x     = (const float*)d_in[0];
    const float* W_ih  = (const float*)d_in[1];
    const float* W_hh  = (const float*)d_in[2];
    const float* b_ih  = (const float*)d_in[3];
    const float* b_hh  = (const float*)d_in[4];
    const float* W_lin = (const float*)d_in[5];
    const float* b_lin = (const float*)d_in[6];
    float* out = (float*)d_out;

    const int B = in_sizes[0] / (TSTEPS * INSZ);   // 2048
    const int nblocks = B / BTILE;                 // 128 persistent workgroups

    lstm_persistent_kernel<<<dim3(nblocks), dim3(256), 0, stream>>>(
        x, W_ih, W_hh, b_ih, b_hh, W_lin, b_lin, out);
}
